// ExtendedSpatialAttention_5884105195654
// MI455X (gfx1250) — matrix operand order repaired and checked by reading, not compiled, not run on hardware
//
#include <hip/hip_runtime.h>
#include <hip/hip_bf16.h>

typedef __bf16 bf16;
typedef __attribute__((ext_vector_type(16))) __bf16 v16bf;
typedef __attribute__((ext_vector_type(8)))  __bf16 bf16x8;
typedef __attribute__((ext_vector_type(8)))  float  v8f;

#define DEV_INLINE __device__ __forceinline__

// ---------------- problem dims ----------------
static constexpr int BB   = 2;     // batch
static constexpr int TT   = 8;     // frames
static constexpr int BT   = 16;    // BB*TT
static constexpr int CC   = 512;   // channels / embed
static constexpr int HWW  = 1024;  // H*W
static constexpr int NH   = 8;     // heads
static constexpr int NTXT = 77;    // context tokens

// ---------------- WMMA fragment helpers (CDNA5 16x16x32 bf16) ----------------
DEV_INLINE v16bf join8(bf16x8 lo, bf16x8 hi) {
  return __builtin_shufflevector(lo, hi, 0,1,2,3,4,5,6,7,8,9,10,11,12,13,14,15);
}

// A fragment: 16(M) x 32(K), row-major source (ld = row stride in elems).
// lane L: m = L&15, hi = L>>4; elems 0..7 = K[hi*8 .. hi*8+7], 8..15 = K[16+hi*8 ..]
DEV_INLINE v16bf load_a_frag(const bf16* base, int ld, int lane) {
  int m = lane & 15, hi = (lane >> 4) & 1;
  const bf16* row = base + m * ld;
  bf16x8 lo = *(const bf16x8*)(row + hi * 8);
  bf16x8 hv = *(const bf16x8*)(row + 16 + hi * 8);
  return join8(lo, hv);
}

// B fragment for A@B^T: 32(K) x 16(N); source row-major (n, k) with row stride ld.
// lane L: n = L&15, hi = L>>4; elems 0..7 = K[hi*8 ..], 8..15 = K[16+hi*8 ..] (matches A).
DEV_INLINE v16bf load_b_frag(const bf16* base, int ld, int lane) {
  int n = lane & 15, hi = (lane >> 4) & 1;
  const bf16* row = base + n * ld + hi * 8;
  return join8(*(const bf16x8*)row, *(const bf16x8*)(row + 16));
}

DEV_INLINE v8f wmma_bf16(v16bf a, v16bf b, v8f c) {
  return __builtin_amdgcn_wmma_f32_16x16x32_bf16(false, a, false, b, (short)0, c, false, false);
}

// Async global->LDS staging (CDNA5 GLOBAL_LOAD_ASYNC_TO_LDS_B128, ASYNCcnt).
// NOTE: per ISA §8 the instruction IOFFSET is added to BOTH the LDS dest and the
// global address, so two ops at offset 0/16 stage 32 contiguous bytes.
DEV_INLINE void async_copy32(void* lds_dst, const void* gsrc) {
  unsigned lds = (unsigned)(size_t)lds_dst;            // generic->LDS: addr[31:0]
  asm volatile("global_load_async_to_lds_b128 %0, %1, off"
               :: "v"(lds), "v"(gsrc) : "memory");
  asm volatile("global_load_async_to_lds_b128 %0, %1, off offset:16"
               :: "v"(lds), "v"(gsrc) : "memory");
}
DEV_INLINE void wait_async0() {
  asm volatile("s_wait_asynccnt 0x0" ::: "memory");
}

// ---------------- f32 -> bf16 convert ----------------
__global__ __launch_bounds__(256) void f2bf_kernel(const float* __restrict__ x,
                                                   bf16* __restrict__ y, int n) {
  for (int i = blockIdx.x * 256 + threadIdx.x; i < n; i += gridDim.x * 256)
    y[i] = (bf16)x[i];
}

// ---------------- GroupNorm1: x(BT,C,HW) f32 -> y(BT,HW,C) f32 (fused transpose) ----
__global__ __launch_bounds__(256)
void groupnorm1_kernel(const float* __restrict__ X, const float* __restrict__ w,
                       const float* __restrict__ b, float* __restrict__ Y) {
  const int GC = 16;                       // channels per group (512/32)
  int bt = blockIdx.x, g = blockIdx.y;
  const float* xg = X + ((long)bt * CC + (long)g * GC) * HWW;  // 16*1024 contiguous
  __shared__ float red[2][8];
  float s = 0.f, ss = 0.f;
  for (int i = threadIdx.x; i < GC * HWW; i += 256) { float v = xg[i]; s += v; ss += v * v; }
  for (int d = 1; d < 32; d <<= 1) { s += __shfl_xor(s, d, 32); ss += __shfl_xor(ss, d, 32); }
  int lane = threadIdx.x & 31, wid = threadIdx.x >> 5;
  if (lane == 0) { red[0][wid] = s; red[1][wid] = ss; }
  __syncthreads();
  float ts = 0.f, tss = 0.f;
  for (int i = 0; i < 8; ++i) { ts += red[0][i]; tss += red[1][i]; }
  float mean = ts / (GC * HWW);
  float rstd = rsqrtf(tss / (GC * HWW) - mean * mean + 1e-5f);
  for (int i = threadIdx.x; i < GC * HWW; i += 256) {
    int ci = i / HWW, hw = i % HWW;
    int c = g * GC + ci;
    Y[((long)bt * HWW + hw) * CC + c] = (xg[i] - mean) * rstd * w[c] + b[c];
  }
}

// ---------------- GroupNorm2: X(BT,HW,C) f32 -> Y(BT,HW,C) f32 ----------------
__global__ __launch_bounds__(256)
void groupnorm2_kernel(const float* __restrict__ X, const float* __restrict__ w,
                       const float* __restrict__ b, float* __restrict__ Y) {
  const int GC = 16;
  int bt = blockIdx.x, g = blockIdx.y;
  const float* xb = X + (long)bt * HWW * CC + g * GC;
  float* yb = Y + (long)bt * HWW * CC + g * GC;
  __shared__ float red[2][8];
  float s = 0.f, ss = 0.f;
  for (int i = threadIdx.x; i < GC * HWW; i += 256) {
    int hw = i >> 4, ci = i & 15;
    float v = xb[(long)hw * CC + ci]; s += v; ss += v * v;
  }
  for (int d = 1; d < 32; d <<= 1) { s += __shfl_xor(s, d, 32); ss += __shfl_xor(ss, d, 32); }
  int lane = threadIdx.x & 31, wid = threadIdx.x >> 5;
  if (lane == 0) { red[0][wid] = s; red[1][wid] = ss; }
  __syncthreads();
  float ts = 0.f, tss = 0.f;
  for (int i = 0; i < 8; ++i) { ts += red[0][i]; tss += red[1][i]; }
  float mean = ts / (GC * HWW);
  float rstd = rsqrtf(tss / (GC * HWW) - mean * mean + 1e-5f);
  for (int i = threadIdx.x; i < GC * HWW; i += 256) {
    int hw = i >> 4, ci = i & 15;
    int c = g * GC + ci;
    yb[(long)hw * CC + ci] = (xb[(long)hw * CC + ci] - mean) * rstd * w[c] + b[c];
  }
}

// ---------------- LayerNorm rows of length Cdim; dual affine outputs (bf16) --------
__global__ __launch_bounds__(256)
void layernorm_dual_kernel(const float* __restrict__ X, int nRows, int Cdim,
                           const float* __restrict__ w1, const float* __restrict__ b1,
                           bf16* __restrict__ out1,
                           const float* __restrict__ w2, const float* __restrict__ b2,
                           bf16* __restrict__ out2) {
  int wid = threadIdx.x >> 5, lane = threadIdx.x & 31;
  int row = blockIdx.x * 8 + wid;
  if (row >= nRows) return;
  const float* x = X + (long)row * Cdim;
  float s = 0.f, ss = 0.f;
  for (int c = lane; c < Cdim; c += 32) { float v = x[c]; s += v; ss += v * v; }
  for (int d = 1; d < 32; d <<= 1) { s += __shfl_xor(s, d, 32); ss += __shfl_xor(ss, d, 32); }
  float mean = s / Cdim;
  float rstd = rsqrtf(ss / Cdim - mean * mean + 1e-5f);
  for (int c = lane; c < Cdim; c += 32) {
    float xn = (x[c] - mean) * rstd;
    out1[(long)row * Cdim + c] = (bf16)(xn * w1[c] + b1[c]);
    if (out2) out2[(long)row * Cdim + c] = (bf16)(xn * w2[c] + b2[c]);
  }
}

// ---------------- GEMM: out = alpha*(A(MxK) @ W(NxK)^T + bias) ----------------
// 256 threads = 8 waves, 128x64 block tile, 32x32 per wave, K-step 32.
// Tiles staged with async global->LDS copies (ASYNCcnt path).
__global__ __launch_bounds__(256)
void gemm_bias_kernel(const bf16* __restrict__ A, long aBatch,
                      const bf16* __restrict__ W, const float* __restrict__ bias,
                      bf16* __restrict__ outB, float* __restrict__ outF, long oBatch,
                      int M, int N, int K, float alpha) {
  __shared__ __align__(16) bf16 As[128 * 32];
  __shared__ __align__(16) bf16 Ws[64 * 32];
  int tid = threadIdx.x, lane = tid & 31, wid = tid >> 5;
  int wm = wid & 3, wn = wid >> 2;
  int bm = blockIdx.x * 128, bn = blockIdx.y * 64;
  const bf16* Ab = A + (long)blockIdx.z * aBatch;
  v8f acc[2][2] = {};
  for (int k0 = 0; k0 < K; k0 += 32) {
    { // stage A tile 128x32 (rows >= M keep stale LDS; those D rows are discarded)
      int row = tid >> 1, seg = (tid & 1) * 16;
      int gr = bm + row;
      if (gr < M) {
        const bf16* src = Ab + (long)gr * K + k0 + seg;
        async_copy32(&As[row * 32 + seg], src);
        __builtin_prefetch(src + 32, 0, 0);   // next K tile
      }
    }
    if (tid < 128) { // stage W tile 64x32 (N,K multiples of 64/32 -> always in-bounds)
      int row = tid >> 1, seg = (tid & 1) * 16;
      async_copy32(&Ws[row * 32 + seg], W + (long)(bn + row) * K + k0 + seg);
    }
    wait_async0();          // own async copies landed in LDS
    __syncthreads();        // everyone's copies visible
    v16bf a0 = load_a_frag(&As[(wm * 32) * 32], 32, lane);
    v16bf a1 = load_a_frag(&As[(wm * 32 + 16) * 32], 32, lane);
    v16bf b0 = load_b_frag(&Ws[(wn * 32) * 32], 32, lane);
    v16bf b1 = load_b_frag(&Ws[(wn * 32 + 16) * 32], 32, lane);
    acc[0][0] = wmma_bf16(a0, b0, acc[0][0]);
    acc[0][1] = wmma_bf16(a0, b1, acc[0][1]);
    acc[1][0] = wmma_bf16(a1, b0, acc[1][0]);
    acc[1][1] = wmma_bf16(a1, b1, acc[1][1]);
    __syncthreads();
  }
  int hi = lane >> 4, nl = lane & 15;
  for (int im = 0; im < 2; ++im)
    for (int in = 0; in < 2; ++in) {
      int ncol = bn + wn * 32 + in * 16 + nl;
      float bv = bias ? bias[ncol] : 0.f;
      #pragma unroll
      for (int r = 0; r < 8; ++r) {
        int mrow = bm + wm * 32 + im * 16 + r + 8 * hi;
        if (mrow < M) {
          float val = (acc[im][in][r] + bv) * alpha;
          long o = (long)blockIdx.z * oBatch + (long)mrow * N + ncol;
          if (outB) outB[o] = (bf16)val;
          if (outF) outF[o] = val;
        }
      }
    }
}

// ---------------- Flash attention: per wave 16 q-rows, 32-key chunks ----------------
// Q,K,V,O layout: (batch, L, E=512) bf16, head h uses cols [h*64, h*64+64).
__global__ __launch_bounds__(256)
void attn_kernel(const bf16* __restrict__ Q, long qBatch,
                 const bf16* __restrict__ Kp, long kBatch,
                 const bf16* __restrict__ Vp, long vBatch,
                 bf16* __restrict__ O, long oBatch,
                 int Lq, int Lk, int kvMod) {
  const int E = 512;
  __shared__ __align__(16) bf16 Kt[32 * 64];        // [key][d]
  __shared__ __align__(16) bf16 Vt[64 * 32];        // [d][key] (transposed)
  __shared__ __align__(16) bf16 Ps[8][16 * 32];     // per-wave P scratch
  int tid = threadIdx.x, lane = tid & 31, wid = tid >> 5;
  int qb = blockIdx.x * 128 + wid * 16;
  int h = blockIdx.y, bz = blockIdx.z;
  const bf16* q  = Q  + (long)bz * qBatch;
  const bf16* kk = Kp + (long)(bz % kvMod) * kBatch;
  const bf16* vv = Vp + (long)(bz % kvMod) * vBatch;
  int m = lane & 15, hi = lane >> 4;

  // Q fragments (16 x 64 -> two 16x32 A-frags), loaded straight from global
  v16bf qa0, qa1;
  {
    const bf16* rq = q + (long)(qb + m) * E + h * 64;
    qa0 = join8(*(const bf16x8*)(rq + hi * 8), *(const bf16x8*)(rq + 16 + hi * 8));
    qa1 = join8(*(const bf16x8*)(rq + 32 + hi * 8), *(const bf16x8*)(rq + 48 + hi * 8));
  }
  float mrun[8], lrun[8];
  #pragma unroll
  for (int r = 0; r < 8; ++r) { mrun[r] = -1e30f; lrun[r] = 0.f; }
  v8f o0 = {}, o1 = {}, o2 = {}, o3 = {};

  int nChunks = (Lk + 31) / 32;
  for (int kc = 0; kc < nChunks; ++kc) {
    { // stage K (row-major) and V (transposed) chunk of 32 keys; zero-fill past Lk
      int key = tid >> 3, dseg = (tid & 7) * 8;
      int gk = kc * 32 + key;
      bf16x8 k8 = {}, v8 = {};
      if (gk < Lk) {
        k8 = *(const bf16x8*)(kk + (long)gk * E + h * 64 + dseg);
        v8 = *(const bf16x8*)(vv + (long)gk * E + h * 64 + dseg);
      }
      *(bf16x8*)(&Kt[key * 64 + dseg]) = k8;
      #pragma unroll
      for (int j = 0; j < 8; ++j) Vt[(dseg + j) * 32 + key] = v8[j];
    }
    __syncthreads();

    // S = Q @ K^T  (two 16x16 key tiles, two d-steps each)
    v8f s0 = {}, s1 = {};
    s0 = wmma_bf16(qa0, load_b_frag(&Kt[0], 64, lane), s0);
    s0 = wmma_bf16(qa1, load_b_frag(&Kt[32], 64, lane), s0);
    s1 = wmma_bf16(qa0, load_b_frag(&Kt[16 * 64], 64, lane), s1);
    s1 = wmma_bf16(qa1, load_b_frag(&Kt[16 * 64 + 32], 64, lane), s1);

    int keyA = kc * 32 + (lane & 15);
    if (keyA >= Lk) {
      #pragma unroll
      for (int r = 0; r < 8; ++r) s0[r] = -1e30f;
    }
    if (keyA + 16 >= Lk) {
      #pragma unroll
      for (int r = 0; r < 8; ++r) s1[r] = -1e30f;
    }

    // online softmax per row (row m = r + 8*hi; 16-lane butterflies stay in-half)
    bf16* pl = &Ps[wid][0];
    #pragma unroll
    for (int r = 0; r < 8; ++r) {
      float a = s0[r], c = s1[r];
      float v = fmaxf(a, c);
      for (int d = 1; d < 16; d <<= 1) v = fmaxf(v, __shfl_xor(v, d, 32));
      float mn = fmaxf(mrun[r], v);
      float corr = __expf(mrun[r] - mn);
      float p0 = __expf(a - mn);
      float p1 = __expf(c - mn);
      float rs = p0 + p1;
      for (int d = 1; d < 16; d <<= 1) rs += __shfl_xor(rs, d, 32);
      lrun[r] = lrun[r] * corr + rs;
      mrun[r] = mn;
      o0[r] *= corr; o1[r] *= corr; o2[r] *= corr; o3[r] *= corr;
      int rowm = r + 8 * hi;
      pl[rowm * 32 + (lane & 15)]      = (bf16)p0;
      pl[rowm * 32 + 16 + (lane & 15)] = (bf16)p1;
    }

    // O += P @ V   (one 16x32 A-frag of P, four 16-wide d tiles of V^T)
    v16bf pa = load_a_frag(pl, 32, lane);
    o0 = wmma_bf16(pa, load_b_frag(&Vt[0], 32, lane), o0);
    o1 = wmma_bf16(pa, load_b_frag(&Vt[16 * 32], 32, lane), o1);
    o2 = wmma_bf16(pa, load_b_frag(&Vt[32 * 32], 32, lane), o2);
    o3 = wmma_bf16(pa, load_b_frag(&Vt[48 * 32], 32, lane), o3);
    __syncthreads();
  }

  #pragma unroll
  for (int r = 0; r < 8; ++r) {
    float inv = 1.0f / lrun[r];
    int row = qb + r + 8 * hi;
    bf16* orow = O + (long)bz * oBatch + (long)row * E + h * 64 + (lane & 15);
    orow[0]  = (bf16)(o0[r] * inv);
    orow[16] = (bf16)(o1[r] * inv);
    orow[32] = (bf16)(o2[r] * inv);
    orow[48] = (bf16)(o3[r] * inv);
  }
}

// ---------------- per-frame residual: xs += vn + gamma * attn_proj ----------------
__global__ __launch_bounds__(256)
void resid_update_kernel(float* __restrict__ xs, const bf16* __restrict__ vn,
                         const float* __restrict__ ap, const float* __restrict__ gamma,
                         long frameOff, long bStrideX, long bStrideA, int n) {
  int b = blockIdx.y;
  float* xp = xs + frameOff + (long)b * bStrideX;
  const bf16* vp = vn + frameOff + (long)b * bStrideX;
  const float* app = ap + (long)b * bStrideA;
  for (int i = blockIdx.x * 256 + threadIdx.x; i < n; i += gridDim.x * 256) {
    int c = i & (CC - 1);
    xp[i] += (float)vp[i] + gamma[c] * app[i];
  }
}

// ---------------- final: out(BT,C,HW) = vn2 + gamma*attn2 (transpose scatter) -------
__global__ __launch_bounds__(256)
void final_combine_kernel(const bf16* __restrict__ vn2, const float* __restrict__ ap2,
                          const float* __restrict__ gamma, float* __restrict__ out) {
  long total = (long)BT * HWW * CC;
  long i = (long)blockIdx.x * 256 + threadIdx.x;
  if (i < total) {
    int c = (int)(i & (CC - 1));
    long rhw = i >> 9;
    int hw = (int)(rhw & (HWW - 1));
    int bt = (int)(rhw >> 10);
    float v = (float)vn2[i] + gamma[c] * ap2[i];
    out[((long)bt * CC + c) * HWW + hw] = v;
  }
}

// =========================== host launcher ===========================
extern "C" void kernel_launch(void* const* d_in, const int* in_sizes, int n_in,
                              void* d_out, int out_size, void* d_ws, size_t ws_size,
                              hipStream_t stream) {
  (void)in_sizes; (void)n_in; (void)out_size; (void)ws_size;
  const float* x        = (const float*)d_in[0];
  const float* context  = (const float*)d_in[1];
  const float* gn1_w    = (const float*)d_in[2];
  const float* gn1_b    = (const float*)d_in[3];
  const float* gn2_w    = (const float*)d_in[4];
  const float* gn2_b    = (const float*)d_in[5];
  const float* sa_lnv_w = (const float*)d_in[6];
  const float* sa_lnv_b = (const float*)d_in[7];
  const float* sa_lnl_w = (const float*)d_in[8];
  const float* sa_lnl_b = (const float*)d_in[9];
  const float* sa_qw    = (const float*)d_in[10];
  const float* sa_qb    = (const float*)d_in[11];
  const float* sa_kw    = (const float*)d_in[12];
  const float* sa_kb    = (const float*)d_in[13];
  const float* sa_vw    = (const float*)d_in[14];
  const float* sa_vb    = (const float*)d_in[15];
  const float* sa_ow    = (const float*)d_in[16];
  const float* sa_ob    = (const float*)d_in[17];
  const float* sa_gamma = (const float*)d_in[18];
  const float* ca_lnv_w = (const float*)d_in[19];
  const float* ca_lnv_b = (const float*)d_in[20];
  const float* ca_lnl_w = (const float*)d_in[21];
  const float* ca_lnl_b = (const float*)d_in[22];
  const float* ca_qw    = (const float*)d_in[23];
  const float* ca_qb    = (const float*)d_in[24];
  const float* ca_kw    = (const float*)d_in[25];
  const float* ca_kb    = (const float*)d_in[26];
  const float* ca_vw    = (const float*)d_in[27];
  const float* ca_vb    = (const float*)d_in[28];
  const float* ca_ow    = (const float*)d_in[29];
  const float* ca_ob    = (const float*)d_in[30];
  const float* ca_gamma = (const float*)d_in[31];
  float* out = (float*)d_out;

  // ---- workspace bump allocator (with aliasing for phase B) ----
  char* ws = (char*)d_ws;
  size_t off = 0;
  auto alloc = [&](size_t bytes) { size_t r = off; off += (bytes + 255) & ~(size_t)255; return r; };
  const long actN  = (long)BT * HWW * CC;            // 8.4M elems
  const long bActN = (long)BB * HWW * CC;            // per-batch activation
  size_t oXS  = alloc(actN * 4);                     // xs (BT,HW,C) f32
  size_t oVN  = alloc(actN * 2);                     // vn bf16
  size_t oLN  = alloc(actN * 2);                     // ln bf16 (contig after vn)
  size_t oQB  = alloc(bActN * 2);                    // q  (B,HW,C) bf16
  size_t oKB  = alloc((long)BB * 2 * HWW * CC * 2);  // k  (B,2HW,C) bf16
  size_t oVB  = alloc((long)BB * 2 * HWW * CC * 2);  // v
  size_t oOB  = alloc(bActN * 2);                    // attn out bf16
  size_t oAP  = alloc(bActN * 4);                    // o-proj f32
  size_t oCTX = alloc((long)BB * NTXT * CC * 2);     // ctx layernorm bf16
  size_t oK2  = alloc((long)BB * NTXT * CC * 2);
  size_t oV2  = alloc((long)BB * NTXT * CC * 2);
  size_t oW   = alloc((size_t)8 * CC * CC * 2);      // bf16 weights

  float* xs   = (float*)(ws + oXS);
  bf16*  vn   = (bf16*)(ws + oVN);
  bf16*  ln   = (bf16*)(ws + oLN);
  bf16*  qb   = (bf16*)(ws + oQB);
  bf16*  kb   = (bf16*)(ws + oKB);
  bf16*  vb   = (bf16*)(ws + oVB);
  bf16*  ob   = (bf16*)(ws + oOB);
  float* ap   = (float*)(ws + oAP);
  bf16*  ctxl = (bf16*)(ws + oCTX);
  bf16*  k2   = (bf16*)(ws + oK2);
  bf16*  v2   = (bf16*)(ws + oV2);
  bf16*  wbf  = (bf16*)(ws + oW);
  // phase-B aliases (lifetimes verified disjoint):
  float* g2   = (float*)(ws + oVN);                  // GN2 out, reuses vn+ln (33.5MB)
  bf16*  vn2  = (bf16*)(ws + oQB);                   // reuses qb..ap (16.77MB)
  bf16*  q2   = (bf16*)(ws + oXS);                   // reuses xs first half
  bf16*  ob2  = (bf16*)(ws + oXS + actN * 2);        // reuses xs second half
  float* ap2  = (float*)(ws + oVN);                  // reuses g2 region after use

  bf16* w_saq = wbf + 0L * CC * CC;
  bf16* w_sak = wbf + 1L * CC * CC;
  bf16* w_sav = wbf + 2L * CC * CC;
  bf16* w_sao = wbf + 3L * CC * CC;
  bf16* w_caq = wbf + 4L * CC * CC;
  bf16* w_cak = wbf + 5L * CC * CC;
  bf16* w_cav = wbf + 6L * CC * CC;
  bf16* w_cao = wbf + 7L * CC * CC;

  const int WN = CC * CC;
  f2bf_kernel<<<1024, 256, 0, stream>>>(sa_qw, w_saq, WN);
  f2bf_kernel<<<1024, 256, 0, stream>>>(sa_kw, w_sak, WN);
  f2bf_kernel<<<1024, 256, 0, stream>>>(sa_vw, w_sav, WN);
  f2bf_kernel<<<1024, 256, 0, stream>>>(sa_ow, w_sao, WN);
  f2bf_kernel<<<1024, 256, 0, stream>>>(ca_qw, w_caq, WN);
  f2bf_kernel<<<1024, 256, 0, stream>>>(ca_kw, w_cak, WN);
  f2bf_kernel<<<1024, 256, 0, stream>>>(ca_vw, w_cav, WN);
  f2bf_kernel<<<1024, 256, 0, stream>>>(ca_ow, w_cao, WN);

  // ---- GN1 + transpose -> xs (BT,HW,C) ----
  groupnorm1_kernel<<<dim3(BT, 32), 256, 0, stream>>>(x, gn1_w, gn1_b, xs);

  // ---- LayerNorm (both affines) for all frames ----
  layernorm_dual_kernel<<<(BT * HWW) / 8, 256, 0, stream>>>(
      xs, BT * HWW, CC, sa_lnv_w, sa_lnv_b, vn, sa_lnl_w, sa_lnl_b, ln);

  const long fStride = (long)TT * HWW * CC;          // per-batch stride in (B,T,HW,C)
  const float qScale = 0.125f;                       // hd^-0.5, hd=64

  // ---- self-attention over frames ----
  for (int idx = 0; idx < TT; ++idx) {
    int Lkv = (idx == 0) ? HWW : 2 * HWW;
    long kvOff = (long)((idx == 0) ? 0 : idx - 1) * HWW * CC;
    // Q = vn[:,idx] @ qw^T * scale
    gemm_bias_kernel<<<dim3(HWW / 128, CC / 64, BB), 256, 0, stream>>>(
        vn + (long)idx * HWW * CC, fStride, w_saq, sa_qb, qb, nullptr,
        (long)HWW * CC, HWW, CC, CC, qScale);
    // K,V from ln rows of frames {idx-1, idx}
    gemm_bias_kernel<<<dim3((Lkv + 127) / 128, CC / 64, BB), 256, 0, stream>>>(
        ln + kvOff, fStride, w_sak, sa_kb, kb, nullptr,
        (long)2 * HWW * CC, Lkv, CC, CC, 1.f);
    gemm_bias_kernel<<<dim3((Lkv + 127) / 128, CC / 64, BB), 256, 0, stream>>>(
        ln + kvOff, fStride, w_sav, sa_vb, vb, nullptr,
        (long)2 * HWW * CC, Lkv, CC, CC, 1.f);
    attn_kernel<<<dim3(HWW / 128, NH, BB), 256, 0, stream>>>(
        qb, (long)HWW * CC, kb, (long)2 * HWW * CC, vb, (long)2 * HWW * CC,
        ob, (long)HWW * CC, HWW, Lkv, BB);
    gemm_bias_kernel<<<dim3(HWW / 128, CC / 64, BB), 256, 0, stream>>>(
        ob, (long)HWW * CC, w_sao, sa_ob, nullptr, ap,
        (long)HWW * CC, HWW, CC, CC, 1.f);
    resid_update_kernel<<<dim3(512, BB), 256, 0, stream>>>(
        xs, vn, ap, sa_gamma, (long)idx * HWW * CC, fStride, (long)HWW * CC, HWW * CC);
  }

  // ---- GN2 (stats over channel-group x HW on (BT,HW,C) layout) ----
  groupnorm2_kernel<<<dim3(BT, 32), 256, 0, stream>>>(xs, gn2_w, gn2_b, g2);

  // ---- cross-attention block ----
  layernorm_dual_kernel<<<(BT * HWW) / 8, 256, 0, stream>>>(
      g2, BT * HWW, CC, ca_lnv_w, ca_lnv_b, vn2, ca_lnv_w, ca_lnv_b, nullptr);
  layernorm_dual_kernel<<<(BB * NTXT + 7) / 8, 256, 0, stream>>>(
      context, BB * NTXT, CC, ca_lnl_w, ca_lnl_b, ctxl, ca_lnl_w, ca_lnl_b, nullptr);

  gemm_bias_kernel<<<dim3(HWW / 128, CC / 64, BT), 256, 0, stream>>>(
      vn2, (long)HWW * CC, w_caq, ca_qb, q2, nullptr, (long)HWW * CC, HWW, CC, CC, qScale);
  gemm_bias_kernel<<<dim3(1, CC / 64, BB), 256, 0, stream>>>(
      ctxl, (long)NTXT * CC, w_cak, ca_kb, k2, nullptr, (long)NTXT * CC, NTXT, CC, CC, 1.f);
  gemm_bias_kernel<<<dim3(1, CC / 64, BB), 256, 0, stream>>>(
      ctxl, (long)NTXT * CC, w_cav, ca_vb, v2, nullptr, (long)NTXT * CC, NTXT, CC, CC, 1.f);
  // each bt pairs with context[bt % 2] (matches jnp.tile in the reference)
  attn_kernel<<<dim3(HWW / 128, NH, BT), 256, 0, stream>>>(
      q2, (long)HWW * CC, k2, (long)NTXT * CC, v2, (long)NTXT * CC,
      ob2, (long)HWW * CC, HWW, NTXT, BB);
  gemm_bias_kernel<<<dim3(HWW / 128, CC / 64, BT), 256, 0, stream>>>(
      ob2, (long)HWW * CC, w_cao, ca_ob, nullptr, ap2, (long)HWW * CC, HWW, CC, CC, 1.f);

  final_combine_kernel<<<(unsigned)((actN + 255) / 256), 256, 0, stream>>>(
      vn2, ap2, ca_gamma, out);
}
